// ActorAgent_69690139345295
// MI455X (gfx1250) — matrix operand order repaired and checked by reading, not compiled, not run on hardware
//
#include <hip/hip_runtime.h>
#include <cstdint>
#include <cstddef>

#define T_LEN     4096
#define BATCH     32
#define OBS_DIM   32
#define D_MODEL   16
#define D_STATE   8
#define NHEADS    8
#define D_INNER   16
#define CONV_DIM  32
#define D_IN_PROJ 56
#define CHUNK     64
#define NCHUNK    (T_LEN / CHUNK)
#define N_ACTIONS 16

typedef _Float16     v16h __attribute__((ext_vector_type(16)));
typedef float        v8f  __attribute__((ext_vector_type(8)));
typedef unsigned int v4u  __attribute__((ext_vector_type(4)));
typedef int          v4i  __attribute__((ext_vector_type(4)));
typedef int          v8i  __attribute__((ext_vector_type(8)));

// ---- LDS layout (float offsets), total 15776 f32 = 63,104 B < 64 KB ----
#define OFF_W     0      // packed small weights (168 used)
#define OFF_STATE 256    // SSM state  [h][p][n] = 8*2*8
#define OFF_CARRY 384    // conv tail (last pre-conv xBC row), 32
#define OFF_Z     416    // z gate, 64x16
#define OFF_DT    1440   // softplus dt, 64x8
#define OFF_ACS   1952   // cumsum(A*dt), 64x8
#define OFF_XBC   2464   // silu(conv(xBC)), 64x32
#define OFF_Y     4512   // y, 64x16
#define OFF_XDT   5536   // x*dt 64x16 (reused as HID in tail)
#define OFF_X1    6560   // relu(obs@W_in^T+b), 64x16
#define OFF_OBS0  7584   // obs ping buffer, 64x32
#define OFF_OBS1  9632   // obs pong buffer, 64x32
#define OFF_ZXG   11680  // zxbcdt 64x64 (stages C/D), then G=CB^T 64x64 (stage E)
#define SMEM_F    15776

// small-weight slots inside OFF_W
#define W_CONVW  0    // 32x2
#define W_CONVB  64
#define W_DTB    96
#define W_NEGA   104
#define W_DSKIP  112
#define W_NORMW  120
#define W_BIN    136
#define W_BOUT   152

// ---------- WMMA fragment helpers (wave32, 16x16x32 f16 -> f32) ----------
// A frag: lane = m (0..15) + 16*kh ; VGPR j pairs per ISA 16-bit A table.
__device__ __forceinline__ v16h load_A(const float* src, int row0, int ld,
                                       int k0, int kmaxAbs) {
  int lane = threadIdx.x & 31;
  int m = lane & 15, kh = lane >> 4;
  v16h a;
#pragma unroll
  for (int j = 0; j < 16; ++j) {
    int i = j >> 1, e = j & 1;
    int k = (i < 4) ? (kh * 8 + 2 * i + e) : (16 + kh * 8 + 2 * (i - 4) + e);
    int kk = k0 + k;
    float v = (kk < kmaxAbs) ? src[(row0 + m) * ld + kk] : 0.0f;
    a[j] = (_Float16)v;
  }
  return a;
}
// B frag for B[k][c] = M[(col0+c)*ld + koff + k]  (i.e. X @ M^T style operand)
__device__ __forceinline__ v16h load_Bt(const float* M, int ld, int col0,
                                        int ncols, int koff, int kmax) {
  int lane = threadIdx.x & 31;
  int c = lane & 15, kh = lane >> 4;
  v16h b;
#pragma unroll
  for (int j = 0; j < 16; ++j) {
    int k = (j < 8) ? (kh * 8 + j) : (16 + kh * 8 + (j - 8));
    float v = 0.0f;
    if (k < kmax && (col0 + c) < ncols) v = M[(col0 + c) * ld + koff + k];
    b[j] = (_Float16)v;
  }
  return b;
}
// D frag store: vgpr i -> row (i + 8*(lane>>4)), col (lane&15)
__device__ __forceinline__ void store_D(float* dst, int row0, int ld, int col0,
                                        v8f d, const float* bias, bool relu) {
  int lane = threadIdx.x & 31;
  int n = lane & 15, half = lane >> 4;
#pragma unroll
  for (int i = 0; i < 8; ++i) {
    float v = d[i];
    if (bias) v += bias[n];
    if (relu) v = fmaxf(v, 0.0f);
    dst[(row0 + i + 8 * half) * ld + col0 + n] = v;
  }
}
__device__ __forceinline__ v8f wmma16(v16h a, v16h b, v8f c) {
  return __builtin_amdgcn_wmma_f32_16x16x32_f16(false, a, false, b, (short)0, c,
                                                false, false);
}

// ---------- TDM async obs-chunk load: 64 rows x 32 f32, row stride 1024 elem ----
__device__ __forceinline__ void async_load_obs(float* lds_dst,
                                               const float* obs, int b, int c) {
#if __has_builtin(__builtin_amdgcn_tensor_load_to_lds)
  uint64_t gaddr =
      (uint64_t)(uintptr_t)(obs + ((size_t)c * CHUNK * BATCH + b) * OBS_DIM);
  uint32_t lds = (uint32_t)(uintptr_t)lds_dst;
  v4u g0;
  g0.x = 1u;                                   // count=1, user descriptor
  g0.y = lds;                                  // lds_addr
  g0.z = (uint32_t)(gaddr & 0xFFFFFFFFu);      // global_addr[31:0]
  g0.w = (uint32_t)((gaddr >> 32) & 0x01FFFFFFu) | (2u << 30);  // type=2
  uint32_t dim0 = OBS_DIM, dim1 = 65536u;
  uint64_t str0 = (uint64_t)BATCH * OBS_DIM;   // 1024 elements between rows
  v8i g1;
  g1[0] = (int)(2u << 16);                     // mask=0, data_size=4B
  g1[1] = (int)((dim0 & 0xFFFFu) << 16);       // dim0[15:0] at bits 63:48
  g1[2] = (int)(((dim0 >> 16) & 0xFFFFu) | ((dim1 & 0xFFFFu) << 16));
  g1[3] = (int)(((dim1 >> 16) & 0xFFFFu) | ((uint32_t)OBS_DIM << 16)); // tile_dim0=32
  g1[4] = (int)(uint32_t)CHUNK;                // tile_dim1=64, tile_dim2=0
  g1[5] = (int)(uint32_t)(str0 & 0xFFFFFFFFu);
  g1[6] = (int)(uint32_t)((str0 >> 32) & 0xFFFFu);  // stride1 = 0
  g1[7] = 0;
  v4i z4 = {0, 0, 0, 0};
#if defined(__clang_major__) && (__clang_major__ >= 23)
  v8i z8 = {0, 0, 0, 0, 0, 0, 0, 0};
  __builtin_amdgcn_tensor_load_to_lds(g0, g1, z4, z4, z8, 0);
#else
  __builtin_amdgcn_tensor_load_to_lds(g0, g1, z4, z4, 0);
#endif
#else
  int lane = threadIdx.x & 31;
  const float* src = obs + ((size_t)c * CHUNK * BATCH + b) * OBS_DIM;
  for (int r = 0; r < CHUNK; ++r)
    lds_dst[r * OBS_DIM + lane] = src[(size_t)r * BATCH * OBS_DIM + lane];
#endif
}
__device__ __forceinline__ void wait_obs() {
#if __has_builtin(__builtin_amdgcn_s_wait_tensorcnt)
  __builtin_amdgcn_s_wait_tensorcnt(0);
#endif
}

__global__ __launch_bounds__(128) void mamba_actor_kernel(
    const float* __restrict__ obs, const float* __restrict__ W_in,
    const float* __restrict__ b_in, const float* __restrict__ in_proj,
    const float* __restrict__ conv_w, const float* __restrict__ conv_b,
    const float* __restrict__ dt_bias, const float* __restrict__ A_log,
    const float* __restrict__ D_skip, const float* __restrict__ norm_w,
    const float* __restrict__ out_proj, const float* __restrict__ W_out,
    const float* __restrict__ b_out, float* __restrict__ out) {
  __shared__ float smem[SMEM_F];
  const int tid = threadIdx.x;
  const int wave = tid >> 5;
  const int b = blockIdx.x;

  float* sw    = smem + OFF_W;
  float* sstat = smem + OFF_STATE;
  float* scar  = smem + OFF_CARRY;
  float* sz    = smem + OFF_Z;
  float* sdt   = smem + OFF_DT;
  float* sacs  = smem + OFF_ACS;
  float* sxbc  = smem + OFF_XBC;
  float* sy    = smem + OFF_Y;
  float* sxdt  = smem + OFF_XDT;   // also HID in tail
  float* sx1   = smem + OFF_X1;
  float* szxg  = smem + OFF_ZXG;   // zxbcdt then G

  // ---- init state / carry / small weights ----
  sstat[tid] = 0.0f;
  if (tid < 32) scar[tid] = 0.0f;
  if (tid < 64) sw[W_CONVW + tid] = conv_w[tid];
  if (tid < 32) sw[W_CONVB + tid] = conv_b[tid];
  if (tid < 8) {
    sw[W_DTB + tid]   = dt_bias[tid];
    sw[W_NEGA + tid]  = -__expf(A_log[tid]);
    sw[W_DSKIP + tid] = D_skip[tid];
  }
  if (tid < 16) {
    sw[W_NORMW + tid] = norm_w[tid];
    sw[W_BIN + tid]   = b_in[tid];
    sw[W_BOUT + tid]  = b_out[tid];
  }
  // resident f16 B-operand fragments (held in VGPRs for the whole sequence)
  v16h bWin = load_Bt(W_in, 32, 0, 16, 0, 32);
  v16h bIP[4];
#pragma unroll
  for (int nt = 0; nt < 4; ++nt)
    bIP[nt] = load_Bt(in_proj, 16, 16 * nt, D_IN_PROJ, 0, 16);
  v16h bOP = load_Bt(out_proj, 16, 0, 16, 0, 16);
  v16h bWO = load_Bt(W_out, 16, 0, 16, 0, 16);

  if (wave == 0) async_load_obs(smem + OFF_OBS0, obs, b, 0);
  __syncthreads();

  for (int c = 0; c < NCHUNK; ++c) {
    float* sobs = smem + ((c & 1) ? OFF_OBS1 : OFF_OBS0);
    if (wave == 0) wait_obs();
    __syncthreads();
    // prefetch next chunk into the other buffer (overlaps the whole chunk)
    if (wave == 0 && c + 1 < NCHUNK)
      async_load_obs(smem + ((c & 1) ? OFF_OBS0 : OFF_OBS1), obs, b, c + 1);

    // ---- stage B: X1 = relu(obs @ W_in^T + b_in)  (64x32)x(32x16) ----
    {
      v16h a = load_A(sobs, wave * 16, OBS_DIM, 0, 32);
      v8f d = {};
      d = wmma16(a, bWin, d);
      store_D(sx1, wave * 16, 16, 0, d, sw + W_BIN, true);
    }
    __syncthreads();

    // ---- stage C: zxbcdt = X1 @ in_proj^T  (64x16)x(16x56->64) ----
    {
      v16h a = load_A(sx1, wave * 16, 16, 0, 16);
#pragma unroll
      for (int nt = 0; nt < 4; ++nt) {
        v8f d = {};
        d = wmma16(a, bIP[nt], d);
        store_D(szxg, wave * 16, 64, 16 * nt, d, nullptr, false);
      }
    }
    __syncthreads();

    // ---- stage D: conv+silu, softplus(dt), z copy ----
    for (int q = tid; q < CHUNK * CONV_DIM; q += 128) {
      int t = q >> 5, j = q & 31;
      float prev = (t > 0) ? szxg[(t - 1) * 64 + 16 + j] : scar[j];
      float cur = szxg[t * 64 + 16 + j];
      float v = sw[W_CONVB + j] + prev * sw[W_CONVW + 2 * j] +
                cur * sw[W_CONVW + 2 * j + 1];
      sxbc[q] = v * (1.0f / (1.0f + __expf(-v)));  // silu
    }
    for (int q = tid; q < CHUNK * NHEADS; q += 128) {
      int t = q >> 3, h = q & 7;
      float raw = szxg[t * 64 + 48 + h] + sw[W_DTB + h];
      sdt[q] = (raw > 20.0f) ? raw : log1pf(__expf(raw));  // softplus
    }
    for (int q = tid; q < CHUNK * D_INNER; q += 128)
      sz[q] = szxg[(q >> 4) * 64 + (q & 15)];
    __syncthreads();

    // carry conv tail; per-head cumsum(A*dt); xdt = x*dt
    if (tid < 32) scar[tid] = szxg[63 * 64 + 16 + tid];
    if (tid < 8) {
      float run = 0.0f, Ah = sw[W_NEGA + tid];
      for (int s = 0; s < CHUNK; ++s) {
        run += Ah * sdt[s * 8 + tid];
        sacs[s * 8 + tid] = run;
      }
    }
    for (int q = tid; q < CHUNK * D_INNER; q += 128) {
      int t = q >> 4, hp = q & 15;
      sxdt[q] = sxbc[t * 32 + hp] * sdt[t * 8 + (hp >> 1)];
    }
    __syncthreads();

    // ---- stage E: G = C @ B^T  (64x8)x(8x64), K padded to 32 ----
    {
      v16h a = load_A(sxbc, wave * 16, CONV_DIM, 24, 32);  // C cols 24..31
#pragma unroll
      for (int st = 0; st < 4; ++st) {
        v16h bb = load_Bt(sxbc, CONV_DIM, 16 * st, 64, 16, 8);  // B cols 16..23
        v8f d = {};
        d = wmma16(a, bb, d);
        store_D(szxg, wave * 16, 64, 16 * st, d, nullptr, false);
      }
    }
    __syncthreads();

    // ---- stage E2: Y_diag + Y_off + D-skip, and chunk-local states ----
    for (int q = tid; q < CHUNK * NHEADS; q += 128) {
      int t = q >> 3, h = q & 7;
      float at = sacs[t * 8 + h];
      float y0 = 0.0f, y1 = 0.0f;
      for (int s = 0; s <= t; ++s) {
        float w = szxg[t * 64 + s] * __expf(at - sacs[s * 8 + h]);
        y0 += w * sxdt[s * 16 + 2 * h];
        y1 += w * sxdt[s * 16 + 2 * h + 1];
      }
      float et = __expf(at);
      float o0 = 0.0f, o1 = 0.0f;
#pragma unroll
      for (int n = 0; n < D_STATE; ++n) {
        float cn = sxbc[t * 32 + 24 + n];
        o0 += cn * sstat[h * 16 + n];
        o1 += cn * sstat[h * 16 + 8 + n];
      }
      float dsk = sw[W_DSKIP + h];
      sy[t * 16 + 2 * h]     = y0 + et * o0 + sxbc[t * 32 + 2 * h] * dsk;
      sy[t * 16 + 2 * h + 1] = y1 + et * o1 + sxbc[t * 32 + 2 * h + 1] * dsk;
    }
    // chunk-local state: 128 outputs, one per thread
    int h2 = tid >> 4, p2 = (tid >> 3) & 1, n2 = tid & 7;
    float acsL = sacs[63 * 8 + h2];
    float acc = 0.0f;
    for (int s = 0; s < CHUNK; ++s)
      acc += sxbc[s * 32 + 16 + n2] * __expf(acsL - sacs[s * 8 + h2]) *
             sxdt[s * 16 + 2 * h2 + p2];
    __syncthreads();
    sstat[tid] = __expf(acsL) * sstat[tid] + acc;  // cross-chunk recurrence
    __syncthreads();

    // ---- stage F: gate with silu(z), RMS-norm ----
    if (tid < CHUNK) {
      float g[16], ss = 0.0f;
#pragma unroll
      for (int j = 0; j < 16; ++j) {
        float zz = sz[tid * 16 + j];
        float gv = sy[tid * 16 + j] * (zz * (1.0f / (1.0f + __expf(-zz))));
        g[j] = gv;
        ss += gv * gv;
      }
      float r = rsqrtf(ss * (1.0f / 16.0f) + 1e-5f);
#pragma unroll
      for (int j = 0; j < 16; ++j) sy[tid * 16 + j] = g[j] * r * sw[W_NORMW + j];
    }
    __syncthreads();

    // ---- hid = relu(y @ out_proj^T) ----
    {
      v16h a = load_A(sy, wave * 16, 16, 0, 16);
      v8f d = {};
      d = wmma16(a, bOP, d);
      store_D(sxdt, wave * 16, 16, 0, d, nullptr, true);  // HID reuses XDT slot
    }
    __syncthreads();

    // ---- logits = hid @ W_out^T + b_out -> global ----
    {
      v16h a = load_A(sxdt, wave * 16, 16, 0, 16);
      v8f d = {};
      d = wmma16(a, bWO, d);
      int lane = tid & 31, n = lane & 15, half = lane >> 4;
#pragma unroll
      for (int i = 0; i < 8; ++i) {
        int t = c * CHUNK + wave * 16 + i + 8 * half;
        out[((size_t)t * BATCH + b) * N_ACTIONS + n] = d[i] + sw[W_BOUT + n];
      }
    }
    __syncthreads();
  }
}

extern "C" void kernel_launch(void* const* d_in, const int* in_sizes, int n_in,
                              void* d_out, int out_size, void* d_ws,
                              size_t ws_size, hipStream_t stream) {
  (void)in_sizes; (void)n_in; (void)out_size; (void)d_ws; (void)ws_size;
  const float* obs      = (const float*)d_in[0];
  const float* W_in     = (const float*)d_in[1];
  const float* b_in     = (const float*)d_in[2];
  const float* in_proj  = (const float*)d_in[3];
  const float* conv_w   = (const float*)d_in[4];
  const float* conv_b   = (const float*)d_in[5];
  const float* dt_bias  = (const float*)d_in[6];
  const float* A_log    = (const float*)d_in[7];
  const float* D_skip   = (const float*)d_in[8];
  const float* norm_w   = (const float*)d_in[9];
  const float* out_proj = (const float*)d_in[10];
  const float* W_out    = (const float*)d_in[11];
  const float* b_out    = (const float*)d_in[12];
  mamba_actor_kernel<<<dim3(BATCH), dim3(128), 0, stream>>>(
      obs, W_in, b_in, in_proj, conv_w, conv_b, dt_bias, A_log, D_skip, norm_w,
      out_proj, W_out, b_out, (float*)d_out);
}
